// GNN2_17471926960716
// MI455X (gfx1250) — matrix operand order repaired and checked by reading, not compiled, not run on hardware
//
#include <hip/hip_runtime.h>
#include <hip/hip_bf16.h>

typedef __attribute__((ext_vector_type(16))) _Float16 v16h;
typedef __attribute__((ext_vector_type(8)))  _Float16 v8h;
typedef __attribute__((ext_vector_type(8)))  float    v8f;
typedef __attribute__((ext_vector_type(4)))  float    f32x4;

#define NDRUG 846
#define EDIM  256
#define NS    128
#define LDA   264            // f16 LDS row stride (halves): 16B aligned, conflict-friendly
#define WROW  260            // staged W row stride (floats): 16B aligned, breaks 1KB-stride conflicts
#define WBUF  (32 * WROW)    // one staged K-slice: 32 rows (floats)

// LDS layout (bytes)
#define OFF_DR    0
#define OFF_H     (NS * LDA * 2)                  // 67584
#define OFF_SCORE (2 * NS * LDA * 2)              // 135168
#define OFF_ATTN  (OFF_SCORE + NS * 4)            // 135680
#define OFF_DEG   (OFF_ATTN + NS * 4)             // 136192  [weighted_ent | d], 512 f32
#define OFF_PEW   (OFF_DEG + 2 * EDIM * 4)        // 138240  512 f32 partials
#define OFF_W     (OFF_PEW + 512 * 4)             // 140288  double-buffered W slices
#define SMEM_BYTES (OFF_W + 2 * WBUF * 4)         // 206848

// ---------------------------------------------------------------------------
// Async global->LDS staging of one 32x256 f32 W slice (32KB) by 512 threads:
// each thread issues four b128 async copies (ASYNCcnt-tracked, no VGPR data).
// ---------------------------------------------------------------------------
__device__ __forceinline__ void stage_w_slice(const float* __restrict__ gsrc,
                                              unsigned lds_base_bytes, int tid)
{
#pragma unroll
    for (int it = 0; it < 4; ++it) {
        int f4  = it * 512 + tid;          // float4 index 0..2047
        int row = f4 >> 4;                 // 16 float4 per 256-float row
        int c4  = f4 & 15;
        const float* g = gsrc + (size_t)row * EDIM + c4 * 4;
        unsigned l = lds_base_bytes + (unsigned)(row * WROW + c4 * 4) * 4u;
        asm volatile("global_load_async_to_lds_b128 %0, %1, off"
                     :: "v"(l), "v"(g) : "memory");
    }
}

__device__ __forceinline__ void wait_async0()
{
#if __has_builtin(__builtin_amdgcn_s_wait_asynccnt)
    __builtin_amdgcn_s_wait_asynccnt(0);
#else
    asm volatile("s_wait_asynccnt 0x0" ::: "memory");
#endif
}

// ---------------------------------------------------------------------------
// Block-cooperative [128x256]x[256x256] pass. 16 waves, wave w owns N-tile w.
// B slices double-buffered in LDS via async copies; A fragments from f16 LDS.
// ---------------------------------------------------------------------------
__device__ __forceinline__ void wmma_pass_async(const _Float16* __restrict__ aS,
                                                const float* __restrict__ Wb,
                                                float* __restrict__ wS, unsigned wbase,
                                                int tid, int lane, int wave, v8f acc[8])
{
    const int m15 = lane & 15;
    const int kb  = (lane < 16) ? 0 : 8;       // ISA 16-bit A-fragment K sub-offset

    stage_w_slice(Wb, wbase, tid);             // prologue: slice k=0 -> buf0
#pragma unroll
    for (int kblk = 0; kblk < 8; ++kblk) {
        wait_async0();                          // this wave's copies landed
        __syncthreads();                        // everyone's copies landed; prev reads done
        const float* cur = wS + (kblk & 1) * WBUF;
        if (kblk < 7)
            stage_w_slice(Wb + (size_t)(kblk + 1) * 32 * EDIM,
                          wbase + (unsigned)(((kblk + 1) & 1) * WBUF * 4), tid);

        // B fragment: lane%16 = N column of this wave, halves = K rows kb..kb+7, 16+kb..16+kb+7
        const float* br = cur + kb * WROW + wave * 16 + m15;
        v16h bf;
#pragma unroll
        for (int i = 0; i < 8; ++i) {
            bf[i]     = (_Float16)br[i * WROW];
            bf[8 + i] = (_Float16)br[(16 + i) * WROW];
        }

        const int k0 = kblk * 32;
        const _Float16* ar0 = aS + (size_t)m15 * LDA + k0 + kb;
        v8h lo = *(const v8h*)(ar0);
        v8h hi = *(const v8h*)(ar0 + 16);
#pragma unroll
        for (int mt = 0; mt < 8; ++mt) {
            v16h af = __builtin_shufflevector(lo, hi,
                      0,1,2,3,4,5,6,7,8,9,10,11,12,13,14,15);
            if (mt < 7) {
                const _Float16* ar = aS + (size_t)((mt + 1) * 16 + m15) * LDA + k0 + kb;
                lo = *(const v8h*)(ar);
                hi = *(const v8h*)(ar + 16);
            }
            acc[mt] = __builtin_amdgcn_wmma_f32_16x16x32_f16(
                          false, af, false, bf, (short)0, acc[mt], false, false);
        }
    }
}

// ---------------------------------------------------------------------------
// Kernel 1: per-drug attention pipeline. One block per drug, 512 threads
// (16 wave32, one WGP). Writes pre-BN activations x[846,256] to workspace.
// ---------------------------------------------------------------------------
__global__ __launch_bounds__(512)
void gnn2_drug_kernel(const int* __restrict__ adj_tail,
                      const int* __restrict__ adj_rel,
                      const float* __restrict__ drug_table,
                      const float* __restrict__ rela_table,
                      const float* __restrict__ ent_table,
                      const float* __restrict__ W1, const float* __restrict__ b1,
                      const float* __restrict__ W2, const float* __restrict__ b2,
                      const float* __restrict__ lin_w, const float* __restrict__ lin_b,
                      float* __restrict__ xbuf)
{
    extern __shared__ char smem[];
    _Float16* drS    = (_Float16*)(smem + OFF_DR);
    _Float16* hS     = (_Float16*)(smem + OFF_H);
    float*    scoreS = (float*)(smem + OFF_SCORE);
    float*    attnS  = (float*)(smem + OFF_ATTN);
    float*    degS   = (float*)(smem + OFF_DEG);
    float*    pewS   = (float*)(smem + OFF_PEW);
    float*    wS     = (float*)(smem + OFF_W);
    const unsigned wbase = (unsigned)(uintptr_t)wS;   // LDS byte address of W buffers

    const int b    = blockIdx.x;
    const int tid  = threadIdx.x;
    const int lane = tid & 31;
    const int wave = tid >> 5;          // 0..15, one N-tile each
    const int ecol = tid & 255;
    const int half = tid >> 8;          // 0/1: row-range splitter for 512 threads

    // --- stage d and dr = d (*) r into LDS (f16) -------------------------
    const float dreg = drug_table[(size_t)b * EDIM + ecol];
    if (tid < 256) degS[EDIM + tid] = dreg;
    if (tid < NS)  scoreS[tid] = 0.0f;
    for (int n = half; n < NS; n += 2) {
        int rel  = adj_rel[(size_t)b * NS + n];
        float rv = rela_table[(size_t)rel * EDIM + ecol];
        drS[n * LDA + ecol] = (_Float16)(dreg * rv);
    }
    __syncthreads();

    const float* W1b = W1 + (size_t)b * EDIM * EDIM;
    const float* W2b = W2 + (size_t)b * EDIM * EDIM;
    const int m15  = lane & 15;
    const int moff = (lane < 16) ? 0 : 8;
    const int ncol = wave * 16 + m15;

    // ---- matmul 1: H = relu(dr @ W1 + b1) -------------------------------
    {
        v8f acc[8] = {};
        wmma_pass_async(drS, W1b, wS, wbase, tid, lane, wave, acc);
#pragma unroll
        for (int mt = 0; mt < 8; ++mt) {
#pragma unroll
            for (int r = 0; r < 8; ++r) {
                int m = mt * 16 + moff + r;
                float v = acc[mt][r] + b1[(size_t)m * EDIM + ncol];
                v = v > 0.0f ? v : 0.0f;
                hS[m * LDA + ncol] = (_Float16)v;
            }
        }
    }
    __syncthreads();

    // ---- matmul 2: h2 = H @ W2 + b2 -> score[n] = sum_f -----------------
    {
        v8f acc[8] = {};
        wmma_pass_async(hS, W2b, wS, wbase, tid, lane, wave, acc);
#pragma unroll
        for (int mt = 0; mt < 8; ++mt) {
#pragma unroll
            for (int r = 0; r < 8; ++r) {
                int m = mt * 16 + moff + r;
                float s = acc[mt][r] + b2[(size_t)m * EDIM + ncol];
                s += __shfl_xor(s, 1, 32);
                s += __shfl_xor(s, 2, 32);
                s += __shfl_xor(s, 4, 32);
                s += __shfl_xor(s, 8, 32);
                if (m15 == 0) atomicAdd(&scoreS[m], s);   // ds_add_f32
            }
        }
    }
    __syncthreads();

    // --- softmax over 128 neighbors --------------------------------------
    if (tid == 0) {
        float mx = scoreS[0];
        for (int n = 1; n < NS; ++n) mx = fmaxf(mx, scoreS[n]);
        float sum = 0.0f;
        for (int n = 0; n < NS; ++n) { float e = expf(scoreS[n] - mx); attnS[n] = e; sum += e; }
        float inv = 1.0f / sum;
        for (int n = 0; n < NS; ++n) attnS[n] *= inv;
    }
    __syncthreads();

    // --- weighted_ent[e] = sum_n attn[n] * ent_table[tail[n]][e] ---------
    float wacc = 0.0f;
    for (int n = half * 64; n < half * 64 + 64; ++n) {
        int tl = adj_tail[(size_t)b * NS + n];
        wacc = fmaf(attnS[n], ent_table[(size_t)tl * EDIM + ecol], wacc);
    }
    pewS[tid] = wacc;
    __syncthreads();
    if (tid < 256) degS[tid] = pewS[tid] + pewS[tid + 256];
    __syncthreads();

    // --- x[f] = relu( [weighted_ent | d] @ lin_w + lin_b ) ---------------
    float part = 0.0f;
    for (int e = half * 256; e < half * 256 + 256; ++e)
        part = fmaf(degS[e], lin_w[(size_t)e * EDIM + ecol], part);
    pewS[tid] = part;
    __syncthreads();
    if (tid < 256) {
        float x = pewS[tid] + pewS[tid + 256] + lin_b[tid];
        x = x > 0.0f ? x : 0.0f;
        xbuf[(size_t)b * EDIM + tid] = x;
    }
}

// ---------------------------------------------------------------------------
// Kernel 2: training-mode BatchNorm over the drug axis. One block per feature.
// ---------------------------------------------------------------------------
__global__ __launch_bounds__(256)
void gnn2_bn_kernel(const float* __restrict__ x,
                    const float* __restrict__ gamma,
                    const float* __restrict__ beta,
                    float* __restrict__ out)
{
    __shared__ float s1[256];
    __shared__ float s2[256];
    const int f = blockIdx.x;
    const int tid = threadIdx.x;

    float a = 0.0f, q = 0.0f;
    for (int d = tid; d < NDRUG; d += 256) {
        float v = x[(size_t)d * EDIM + f];
        a += v; q += v * v;
    }
    s1[tid] = a; s2[tid] = q;
    __syncthreads();
    for (int st = 128; st > 0; st >>= 1) {
        if (tid < st) { s1[tid] += s1[tid + st]; s2[tid] += s2[tid + st]; }
        __syncthreads();
    }
    const float invN  = 1.0f / (float)NDRUG;
    const float mean  = s1[0] * invN;
    const float var   = s2[0] * invN - mean * mean;
    const float scale = rsqrtf(var + 1e-5f) * gamma[f];
    const float shift = beta[f] - mean * scale;
    for (int d = tid; d < NDRUG; d += 256)
        out[(size_t)d * EDIM + f] = x[(size_t)d * EDIM + f] * scale + shift;
}

// ---------------------------------------------------------------------------
// Kernel 3: pass-through outputs 2 and 3 of the reference tuple.
// ---------------------------------------------------------------------------
__global__ __launch_bounds__(256)
void gnn2_tail_kernel(const float* __restrict__ gnn1,
                      const int* __restrict__ idx,
                      float* __restrict__ out)
{
    const int i = blockIdx.x * 256 + threadIdx.x;
    const int n = NDRUG * EDIM;
    if (i < n) out[n + i] = gnn1[i];
    if (i == 0) out[2 * n] = (float)idx[0];
}

extern "C" void kernel_launch(void* const* d_in, const int* in_sizes, int n_in,
                              void* d_out, int out_size, void* d_ws, size_t ws_size,
                              hipStream_t stream)
{
    const float* gnn1  = (const float*)d_in[0];
    const int*   idx   = (const int*)d_in[1];
    // d_in[2] = drug_name (arange; row index used directly)
    const int*   atail = (const int*)d_in[3];
    const int*   arel  = (const int*)d_in[4];
    const float* dtab  = (const float*)d_in[5];
    const float* rtab  = (const float*)d_in[6];
    const float* etab  = (const float*)d_in[7];
    const float* W1    = (const float*)d_in[8];
    const float* b1    = (const float*)d_in[9];
    const float* W2    = (const float*)d_in[10];
    const float* b2    = (const float*)d_in[11];
    const float* linw  = (const float*)d_in[12];
    const float* linb  = (const float*)d_in[13];
    const float* gamma = (const float*)d_in[14];
    const float* beta  = (const float*)d_in[15];

    float* out  = (float*)d_out;
    float* xbuf = (float*)d_ws;   // [846,256] pre-BN activations

    gnn2_drug_kernel<<<NDRUG, 512, SMEM_BYTES, stream>>>(
        atail, arel, dtab, rtab, etab, W1, b1, W2, b2, linw, linb, xbuf);
    gnn2_bn_kernel<<<EDIM, 256, 0, stream>>>(xbuf, gamma, beta, out);
    gnn2_tail_kernel<<<(NDRUG * EDIM + 255) / 256, 256, 0, stream>>>(gnn1, idx, out);
}
